// HGRN2Attention_30743375905058
// MI455X (gfx1250) — matrix operand order repaired and checked by reading, not compiled, not run on hardware
//
#include <hip/hip_runtime.h>
#include <hip/hip_bf16.h>
#include <math.h>

typedef __bf16 bf16;
typedef __attribute__((ext_vector_type(16))) __bf16 v16bf;
typedef __attribute__((ext_vector_type(8)))  float  v8f;
typedef __attribute__((ext_vector_type(4)))  unsigned int v4u;
typedef __attribute__((ext_vector_type(8)))  int v8i;
typedef __attribute__((ext_vector_type(4)))  int v4i;

#define B_SZ   4
#define L_SZ   4096
#define D_SZ   2048
#define H_SZ   16
#define DK_SZ  128
#define DV_SZ  128
#define CK     64
#define SCALE_F 0.08838834764831845f   /* 128^-0.5 */
#define EPS_F   1e-5f

#if defined(__has_builtin)
# if __has_builtin(__builtin_amdgcn_tensor_load_to_lds) && __has_builtin(__builtin_amdgcn_s_wait_tensorcnt)
#  define HAVE_TDM 1
# endif
#endif
#ifndef HAVE_TDM
# define HAVE_TDM 0
#endif

static __device__ __forceinline__ v8f wmma_bf16(const v16bf& a, const v16bf& b, const v8f& c) {
  return __builtin_amdgcn_wmma_f32_16x16x32_bf16(false, a, false, b, (short)0, c, false, false);
}

// ---------------------------------------------------------------- cvt f32->bf16
__global__ void cvt_f32_bf16(const float* __restrict__ in, bf16* __restrict__ out, size_t n) {
  size_t i = (size_t)blockIdx.x * blockDim.x + threadIdx.x;
  if (i < n) out[i] = (bf16)in[i];
}

// ---------------------------------------------------------------- WMMA GEMM
// C[M,N] = A[M,K] @ B[K,N], bf16 in, fp32 accumulate. Block tile 128x128,
// 8 waves (4x2), each wave 2x4 16x16 tiles, K-step 32, double-buffered LDS,
// register prefetch of the next K-tile, B stored K-transposed in LDS so
// both A and B fragments are contiguous 16B ds_load_b128.
enum { EPI_F32 = 0, EPI_SILU = 1, EPI_GATE = 2, EPI_BF16 = 3 };

template <int EPI>
__global__ __launch_bounds__(256)
void gemm_bf16_kernel(const bf16* __restrict__ A, const bf16* __restrict__ Bm,
                      void* __restrict__ out0, void* __restrict__ out1,
                      int M, int N, int K)
{
  __shared__ __align__(16) bf16 As[2][128][32];
  __shared__ __align__(16) bf16 BsT[2][128][40];   // [n][k], pad 8 -> 80B rows

  const int tid  = threadIdx.x;
  const int lane = tid & 31;
  const int wave = tid >> 5;
  const int wr   = wave & 3;       // 0..3 : 32-row slab
  const int wc   = wave >> 2;      // 0..1 : 64-col slab
  const int m    = lane & 15;
  const int half = lane >> 4;
  const int bm0  = blockIdx.x * 128;
  const int bn0  = blockIdx.y * 128;
  const int nk   = K / 32;

  // per-thread staging coordinates: 2 16B vectors for A, 2 for B
  const int ar0 = tid >> 2,          ac0 = (tid & 3) * 8;
  const int ar1 = (tid + 256) >> 2,  ac1 = ((tid + 256) & 3) * 8;
  const int br0 = tid >> 4,          bc0 = (tid & 15) * 8;
  const int br1 = (tid + 256) >> 4,  bc1 = ((tid + 256) & 15) * 8;

  v8f acc[2][4];
#pragma unroll
  for (int r = 0; r < 2; ++r)
#pragma unroll
    for (int c = 0; c < 4; ++c) acc[r][c] = (v8f){0,0,0,0,0,0,0,0};

  uint4 pa0, pa1, pb0, pb1;
  // prologue: tile 0
  pa0 = *(const uint4*)&A[(size_t)(bm0 + ar0) * K + ac0];
  pa1 = *(const uint4*)&A[(size_t)(bm0 + ar1) * K + ac1];
  pb0 = *(const uint4*)&Bm[(size_t)br0 * N + bn0 + bc0];
  pb1 = *(const uint4*)&Bm[(size_t)br1 * N + bn0 + bc1];
  {
    *(uint4*)&As[0][ar0][ac0] = pa0;
    *(uint4*)&As[0][ar1][ac1] = pa1;
    const bf16* p0 = (const bf16*)&pb0;
    const bf16* p1 = (const bf16*)&pb1;
#pragma unroll
    for (int u = 0; u < 8; ++u) BsT[0][bc0 + u][br0] = p0[u];
#pragma unroll
    for (int u = 0; u < 8; ++u) BsT[0][bc1 + u][br1] = p1[u];
  }
  __syncthreads();

  for (int kt = 0; kt < nk; ++kt) {
    const int cur = kt & 1;
    if (kt + 1 < nk) {          // prefetch next tile into registers (in flight)
      const int k0 = (kt + 1) * 32;
      pa0 = *(const uint4*)&A[(size_t)(bm0 + ar0) * K + k0 + ac0];
      pa1 = *(const uint4*)&A[(size_t)(bm0 + ar1) * K + k0 + ac1];
      pb0 = *(const uint4*)&Bm[(size_t)(k0 + br0) * N + bn0 + bc0];
      pb1 = *(const uint4*)&Bm[(size_t)(k0 + br1) * N + bn0 + bc1];
    }

    v16bf af[2], bfv[4];
#pragma unroll
    for (int r = 0; r < 2; ++r) {
      int mrow = wr * 32 + r * 16 + m;
#pragma unroll
      for (int v = 0; v < 8; ++v) {
        int kk = ((v >> 2) << 4) + (half << 3) + ((v & 3) << 1);
        af[r][2 * v]     = As[cur][mrow][kk];
        af[r][2 * v + 1] = As[cur][mrow][kk + 1];
      }
    }
#pragma unroll
    for (int c = 0; c < 4; ++c) {
      int ncol = wc * 64 + c * 16 + m;
#pragma unroll
      for (int v = 0; v < 8; ++v) {
        int kk = ((v >> 2) << 4) + (half << 3) + ((v & 3) << 1);
        bfv[c][2 * v]     = BsT[cur][ncol][kk];      // table order in k
        bfv[c][2 * v + 1] = BsT[cur][ncol][kk + 1];
      }
    }
#pragma unroll
    for (int r = 0; r < 2; ++r)
#pragma unroll
      for (int c = 0; c < 4; ++c)
        acc[r][c] = wmma_bf16(af[r], bfv[c], acc[r][c]);

    if (kt + 1 < nk) {          // commit prefetched tile to the other buffer
      *(uint4*)&As[cur ^ 1][ar0][ac0] = pa0;
      *(uint4*)&As[cur ^ 1][ar1][ac1] = pa1;
      const bf16* p0 = (const bf16*)&pb0;
      const bf16* p1 = (const bf16*)&pb1;
#pragma unroll
      for (int u = 0; u < 8; ++u) BsT[cur ^ 1][bc0 + u][br0] = p0[u];
#pragma unroll
      for (int u = 0; u < 8; ++u) BsT[cur ^ 1][bc1 + u][br1] = p1[u];
    }
    __syncthreads();
  }

  // epilogue
#pragma unroll
  for (int r = 0; r < 2; ++r) {
#pragma unroll
    for (int c = 0; c < 4; ++c) {
#pragma unroll
      for (int e = 0; e < 8; ++e) {
        int row = bm0 + wr * 32 + r * 16 + e + 8 * half;
        int col = bn0 + wc * 64 + c * 16 + m;
        size_t idx = (size_t)row * N + col;
        float x = acc[r][c][e];
        if (EPI == EPI_F32) {
          ((float*)out0)[idx] = x;
        } else if (EPI == EPI_SILU) {
          ((bf16*)out0)[idx] = (bf16)(x / (1.f + __expf(-x)));
        } else if (EPI == EPI_GATE) {
          float s = 1.f / (1.f + __expf(-x));
          ((bf16*)out0)[idx]  = (bf16)(1.f - s);                           // k
          ((float*)out1)[idx] = fminf(x, 0.f) - log1pf(__expf(-fabsf(x))); // g
        } else { // EPI_BF16
          ((bf16*)out0)[idx] = (bf16)x;
        }
      }
    }
  }
}

// ---------------------------------------------------------------- TDM v-tile load
// D# for a 2D tile: 64 rows x 64 cols of 2-byte elements out of a row-major
// [B*L, D] tensor (row stride D), loaded to LDS. Groups per CDNA5 ISA.
// 6-arg builtin: (v4u g0, v8i g1, v4i g2, v4i g3, v8i g4, i32 cpol).
#if HAVE_TDM
static __device__ __forceinline__ void tdm_load_v_tile(unsigned lds_off,
                                                       unsigned long long gaddr)
{
  v4u g0;
  g0[0] = 1u;                                                   // count=1
  g0[1] = lds_off;                                              // lds_addr
  g0[2] = (unsigned)(gaddr & 0xffffffffu);                      // global_addr lo
  g0[3] = (unsigned)((gaddr >> 32) & 0x01ffffffu) | 0x80000000u; // addr hi | type=2
  v8i g1;
  g1[0] = 0x00010000;                       // wg_mask=0, data_size=1 (2 bytes)
  g1[1] = (int)((unsigned)(D_SZ & 0xffff) << 16);          // tensor_dim0 lo16
  g1[2] = (int)(((unsigned)D_SZ >> 16) |
                ((unsigned)((B_SZ * L_SZ) & 0xffff) << 16)); // dim0 hi | dim1 lo
  g1[3] = (int)(((unsigned)(B_SZ * L_SZ) >> 16) |
                (64u << 16));                              // dim1 hi | tile_dim0=64
  g1[4] = 64;                               // tile_dim1=64, tile_dim2=0
  g1[5] = D_SZ;                             // tensor_dim0_stride lo32 (elements)
  g1[6] = 0;                                // stride hi | dim1_stride lo
  g1[7] = 0;
  v4i z4 = (v4i){0, 0, 0, 0};
  v8i z8 = (v8i){0, 0, 0, 0, 0, 0, 0, 0};
  __builtin_amdgcn_tensor_load_to_lds(g0, g1, z4, z4, z8, 0);
}
#endif

// ---------------------------------------------------------------- GLA chunk scan
// grid = (B*H, DV/64); block = 128 threads (4 waves); each wave owns a
// 16-column slab of the 128x128 fp32 state S in registers (8 v8f tiles).
__device__ __forceinline__ float getS(const v8f& T, int kloc, int half) {
  float own   = T[kloc & 7];
  float other = __shfl_xor(own, 16, 32);
  return ((kloc >> 3) == half) ? own : other;
}

__global__ __launch_bounds__(128)
void gla_kernel(const bf16* __restrict__ qb, const bf16* __restrict__ kb,
                const bf16* __restrict__ vb, const float* __restrict__ gf,
                float* __restrict__ og)
{
  __shared__ __align__(16) bf16 qe[CK][DK_SZ];    // 16 KB
  __shared__ __align__(16) bf16 kd[CK][DK_SZ];    // 16 KB, reused as v tile
  __shared__ __align__(16) bf16 kuT[DK_SZ][72];   // 18 KB, [d][t], padded rows
  __shared__ __align__(16) bf16 Abuf[CK][CK];     //  8 KB
  __shared__ float blast[DK_SZ];                  // 512 B

  bf16 (*vsh)[64] = (bf16(*)[64])&kd[0][0];   // alias: kd dead after phase 3

  const int bh = blockIdx.x;
  const int b  = bh / H_SZ, h = bh % H_SZ;
  const int D  = H_SZ * DK_SZ;
  const int tid = threadIdx.x, lane = tid & 31, wave = tid >> 5;
  const int m = lane & 15, half = lane >> 4;
  const int y64 = (int)blockIdx.y * 64;              // dv slab of this block
  const int ew  = y64 + wave * 16;                   // this wave's dv columns
  const int vcl = wave * 16;                         // local col in vsh

  v8f S[8];
#pragma unroll
  for (int j = 0; j < 8; ++j) S[j] = (v8f){0,0,0,0,0,0,0,0};

  const size_t base = ((size_t)b * L_SZ) * D + (size_t)h * DK_SZ;

  for (int ch = 0; ch < L_SZ / CK; ++ch) {
    const size_t cb = base + (size_t)ch * CK * D;

    // phase 1: per-d cumsum of g -> qe = q*exp(b)*scale, kd = k*exp(-b)
    {
      int d = tid;  // 128 threads == DK
      float bs = 0.f;
      for (int t = 0; t < CK; ++t) {
        size_t idx = cb + (size_t)t * D + d;
        bs += gf[idx];
        qe[t][d] = (bf16)((float)qb[idx] * __expf(bs) * SCALE_F);
        kd[t][d] = (bf16)((float)kb[idx] * __expf(-bs));
      }
      blast[d] = bs;
    }
    __syncthreads();

    // phase 2: kuT[d][t] = k * exp(b_last - b)   (row-contiguous per thread)
    {
      int d = tid;
      float bs = 0.f, bl = blast[d];
      for (int t = 0; t < CK; ++t) {
        size_t idx = cb + (size_t)t * D + d;
        bs += gf[idx];
        kuT[d][t] = (bf16)((float)kb[idx] * __expf(bl - bs));
      }
    }
    __syncthreads();

    // phase 3: A = tril(qe @ kd^T), 4x4 tiles over 4 waves, K=128
    for (int p = wave; p < 16; p += 4) {
      int ti = p >> 2, si = p & 3;
      v8f c = (v8f){0,0,0,0,0,0,0,0};
      if (si <= ti) {
#pragma unroll
        for (int ks = 0; ks < 4; ++ks) {
          int k0 = ks * 32;
          v16bf af, bfv;
#pragma unroll
          for (int v = 0; v < 8; ++v) {
            int kk = ((v >> 2) << 4) + (half << 3) + ((v & 3) << 1);
            af[2 * v]     = qe[ti * 16 + m][k0 + kk];
            af[2 * v + 1] = qe[ti * 16 + m][k0 + kk + 1];
          }
#pragma unroll
          for (int v = 0; v < 8; ++v) {
            int kk = ((v >> 2) << 4) + (half << 3) + ((v & 3) << 1);           // B[k=d, n=s] = kd[s][d]
            bfv[2 * v]     = kd[si * 16 + m][k0 + kk];
            bfv[2 * v + 1] = kd[si * 16 + m][k0 + kk + 1];
          }
          c = wmma_bf16(af, bfv, c);
        }
      }
#pragma unroll
      for (int e = 0; e < 8; ++e) {
        int t = ti * 16 + e + 8 * half;
        int s = si * 16 + m;
        Abuf[t][s] = (bf16)((s <= t) ? c[e] : 0.f);
      }
    }
    __syncthreads();   // kd now dead; its LDS becomes the v tile

    // stage v tile (64 t x 64 dv cols of this block) into LDS via TDM
#if HAVE_TDM
    if (wave == 0) {
      unsigned lds_off = (unsigned)(uintptr_t)&vsh[0][0];
      unsigned long long gaddr =
          (unsigned long long)(uintptr_t)vb + ((unsigned long long)(cb + y64)) * 2ull;
      tdm_load_v_tile(lds_off, gaddr);
      __builtin_amdgcn_s_wait_tensorcnt(0);
    }
#else
#pragma unroll
    for (int i = tid; i < 512; i += 128) {             // 64x64 bf16, 16B vectors
      int t = i >> 3, col = (i & 7) * 8;
      *(uint4*)&vsh[t][col] = *(const uint4*)&vb[cb + (size_t)t * D + y64 + col];
    }
#endif
    __syncthreads();

    // hoisted fragments: v B-frags depend only on ks; S B-frags only on j.
    v16bf vfrag[2];
#pragma unroll
    for (int ks = 0; ks < 2; ++ks) {
#pragma unroll
      for (int v = 0; v < 8; ++v) {
        int s = ks * 32 + ((v >> 2) << 4) + (half << 3) + ((v & 3) << 1);
        vfrag[ks][2 * v]     = vsh[s][vcl + m];
        vfrag[ks][2 * v + 1] = vsh[s + 1][vcl + m];
      }
    }
    v16bf sfrag[4];
#pragma unroll
    for (int j = 0; j < 4; ++j) {
#pragma unroll
      for (int v = 0; v < 8; ++v) {
        float lo0 = getS(S[2 * j],     2 * v,     half);
        float lo1 = getS(S[2 * j],     2 * v + 1, half);
        float hi0 = getS(S[2 * j + 1], 2 * v,     half);
        float hi1 = getS(S[2 * j + 1], 2 * v + 1, half);
        sfrag[j][2 * v]     = (bf16)(half ? hi0 : lo0);
        sfrag[j][2 * v + 1] = (bf16)(half ? hi1 : lo1);
      }
    }

    // phase 4: o = A @ v + qe @ S  (wave-local columns ew..ew+15)
    for (int ti = 0; ti < 4; ++ti) {
      v8f c = (v8f){0,0,0,0,0,0,0,0};
#pragma unroll
      for (int ks = 0; ks < 2; ++ks) {               // A @ v, K = 64
        int k0 = ks * 32;
        v16bf af;
#pragma unroll
        for (int v = 0; v < 8; ++v) {
          int kk = ((v >> 2) << 4) + (half << 3) + ((v & 3) << 1);
          af[2 * v]     = Abuf[ti * 16 + m][k0 + kk];
          af[2 * v + 1] = Abuf[ti * 16 + m][k0 + kk + 1];
        }
        c = wmma_bf16(af, vfrag[ks], c);
      }
#pragma unroll
      for (int j = 0; j < 4; ++j) {                  // qe @ S, K = 128
        int k0 = j * 32;
        v16bf af;
#pragma unroll
        for (int v = 0; v < 8; ++v) {
          int kk = (half << 4) + (v << 1);
          af[2 * v]     = qe[ti * 16 + m][k0 + kk];
          af[2 * v + 1] = qe[ti * 16 + m][k0 + kk + 1];
        }
        c = wmma_bf16(af, sfrag[j], c);
      }
#pragma unroll
      for (int e = 0; e < 8; ++e) {
        int t = ti * 16 + e + 8 * half;
        og[cb + (size_t)t * D + ew + m] = c[e];
      }
    }

    // phase 5: S = exp(b_last) * S + ku^T @ v   (wave-local columns)
#pragma unroll
    for (int j = 0; j < 8; ++j) {
      v8f c = S[j];
#pragma unroll
      for (int r = 0; r < 8; ++r) {
        int d = j * 16 + r + 8 * half;
        c[r] *= __expf(blast[d]);
      }
#pragma unroll
      for (int ks = 0; ks < 2; ++ks) {               // K = 64 over t
        int k0 = ks * 32;
        v16bf af;
#pragma unroll
        for (int v = 0; v < 8; ++v) {
          int kk = ((v >> 2) << 4) + (half << 3) + ((v & 3) << 1);
          af[2 * v]     = kuT[j * 16 + m][k0 + kk];  // A[m=d, k=t] = kuT[d][t]
          af[2 * v + 1] = kuT[j * 16 + m][k0 + kk + 1];
        }
        c = wmma_bf16(af, vfrag[ks], c);
      }
      S[j] = c;
    }
    __syncthreads();
  }
}

// ---------------------------------------------------------------- RMS norm
__global__ __launch_bounds__(256)
void rmsnorm_kernel(const float* __restrict__ o, const float* __restrict__ gw,
                    bf16* __restrict__ out, int D)
{
  __shared__ float red[256];
  int row = blockIdx.x, tid = threadIdx.x;
  const float* p = o + (size_t)row * D;
  float s = 0.f;
  for (int i = tid; i < D; i += 256) { float x = p[i]; s += x * x; }
  red[tid] = s; __syncthreads();
  for (int off = 128; off > 0; off >>= 1) {
    if (tid < off) red[tid] += red[tid + off];
    __syncthreads();
  }
  float inv = rsqrtf(red[0] / (float)D + EPS_F);
  for (int i = tid; i < D; i += 256)
    out[(size_t)row * D + i] = (bf16)(p[i] * inv * gw[i]);
}

// ---------------------------------------------------------------- launch
extern "C" void kernel_launch(void* const* d_in, const int* in_sizes, int n_in,
                              void* d_out, int out_size, void* d_ws, size_t ws_size,
                              hipStream_t stream)
{
  const float* x   = (const float*)d_in[0];
  const float* Wq  = (const float*)d_in[1];
  const float* Wf  = (const float*)d_in[2];
  const float* Wi  = (const float*)d_in[3];
  const float* Wo  = (const float*)d_in[4];
  const float* gw  = (const float*)d_in[5];

  const size_t NX = (size_t)B_SZ * L_SZ * D_SZ;     // 33.5M elems
  const size_t NW = (size_t)D_SZ * D_SZ;            // 4.2M elems

  char* ws = (char*)d_ws;
  size_t off = 0;
  bf16* xbf  = (bf16*)(ws + off); off += NX * 2;
  bf16* wqb  = (bf16*)(ws + off); off += NW * 2;
  bf16* wfb  = (bf16*)(ws + off); off += NW * 2;
  bf16* wib  = (bf16*)(ws + off); off += NW * 2;
  bf16* wob  = (bf16*)(ws + off); off += NW * 2;
  bf16* qbf  = (bf16*)(ws + off); off += NX * 2;
  bf16* kbf  = (bf16*)(ws + off); off += NX * 2;
  bf16* vbf  = (bf16*)(ws + off); off += NX * 2;
  float* gws = (float*)(ws + off); off += NX * 4;
  float* ows = (float*)(ws + off); off += NX * 4;
  bf16* onrm = (bf16*)(ws + off); off += NX * 2;

  cvt_f32_bf16<<<(unsigned)((NX + 255) / 256), 256, 0, stream>>>(x,  xbf, NX);
  cvt_f32_bf16<<<(unsigned)((NW + 255) / 256), 256, 0, stream>>>(Wq, wqb, NW);
  cvt_f32_bf16<<<(unsigned)((NW + 255) / 256), 256, 0, stream>>>(Wf, wfb, NW);
  cvt_f32_bf16<<<(unsigned)((NW + 255) / 256), 256, 0, stream>>>(Wi, wib, NW);
  cvt_f32_bf16<<<(unsigned)((NW + 255) / 256), 256, 0, stream>>>(Wo, wob, NW);

  const int M = B_SZ * L_SZ;            // 16384
  dim3 gg(M / 128, D_SZ / 128);         // 128 x 16 blocks

  gemm_bf16_kernel<EPI_SILU><<<gg, 256, 0, stream>>>(xbf, wqb, qbf, nullptr, M, D_SZ, D_SZ);
  gemm_bf16_kernel<EPI_GATE><<<gg, 256, 0, stream>>>(xbf, wfb, kbf, gws,     M, D_SZ, D_SZ);
  gemm_bf16_kernel<EPI_BF16><<<gg, 256, 0, stream>>>(xbf, wib, vbf, nullptr, M, D_SZ, D_SZ);

  gla_kernel<<<dim3(B_SZ * H_SZ, DV_SZ / 64), 128, 0, stream>>>(qbf, kbf, vbf, gws, ows);

  rmsnorm_kernel<<<M, 256, 0, stream>>>(ows, gw, onrm, D_SZ);

  gemm_bf16_kernel<EPI_F32><<<gg, 256, 0, stream>>>(onrm, wob, d_out, nullptr, M, D_SZ, D_SZ);
}
